// SelfAttn_86096914416008
// MI455X (gfx1250) — compile-verified, matrix operand order checked by reading, not run on hardware
//
#include <hip/hip_runtime.h>

// ---------------------------------------------------------------------------
// Single-head self-attention on MI455X (gfx1250, wave32, WMMA bf16)
//   B=8, S=2048, E=768, fp32 in/out, bf16 WMMA with f32 accumulation.
// Pipeline:
//   0) prep     : x -> bf16 (xb); Wq/Wk/Wv/Wo -> bf16 transposed (Wt[n][k])
//   1) qkv_proj : xb @ Wt{q,k,v} + bias -> Q,K (bf16 row-major), Vt ([b][e][s])
//                 register-blocked: each wave owns a 16x64 output strip
//   2) flash    : streaming-softmax attention, ctx (bf16 row-major)
//   3) out_proj : ctx @ Wto + bo -> fp32 d_out (16x64 strip per wave)
// All GEMM operands are bf16 and K-contiguous, so every fragment load is two
// 16-byte chunks per lane (global_load_b128 / ds_load_b128).
// Workspace: 5*B*S*E + 4*E*E bf16 ~ 130 MB (L2 is 192 MB).
// ---------------------------------------------------------------------------

#define Bq 8
#define Sq 2048
#define Eq 768
#define EE (Eq * Eq)

typedef __attribute__((ext_vector_type(16))) __bf16 v16bf;
typedef __attribute__((ext_vector_type(8)))  float  v8f;

__device__ __forceinline__ __bf16 f2bf(float x) {
  unsigned u = __float_as_uint(x);
  unsigned r = u + 0x7FFFu + ((u >> 16) & 1u);   // round-to-nearest-even
  unsigned short s = (unsigned short)(r >> 16);
  return __builtin_bit_cast(__bf16, s);
}

// Fragment loader for bf16 K-contiguous tiles (A: M x K row-major, or
// B stored K-major as N x K). `tile` points at element [0][k0]; ld even.
// Lane: r = lane&15, hi = lane>>4. Two contiguous 16B chunks per lane.
__device__ __forceinline__ v16bf ld_frag_bf16(const __bf16* tile, int ld,
                                              int r, int hi) {
  union { v16bf v; unsigned u[8]; } f;
  const unsigned* p = reinterpret_cast<const unsigned*>(tile + (size_t)r * ld);
  int o = hi * 4;
#pragma unroll
  for (int j = 0; j < 4; ++j) f.u[j]     = p[o + j];
#pragma unroll
  for (int j = 0; j < 4; ++j) f.u[4 + j] = p[8 + o + j];
  return f.v;
}

// ---------------------------------------------------------------------------
// Kernel 0: one-time precision/layout prep.
//   idx < BSE            : xb[idx] = bf16(x[idx])
//   idx in [BSE, BSE+4EE): Wt_m[n*E + k] = bf16(W_m[k*E + n])   (transpose)
// ---------------------------------------------------------------------------
__global__ void prep_kernel(const float* __restrict__ x,
                            const float* __restrict__ Wqm,
                            const float* __restrict__ Wkm,
                            const float* __restrict__ Wvm,
                            const float* __restrict__ Wom,
                            __bf16* __restrict__ xb,
                            __bf16* __restrict__ Wt) {  // 4 matrices packed
  const size_t BSE = (size_t)Bq * Sq * Eq;
  size_t idx = (size_t)blockIdx.x * blockDim.x + threadIdx.x;
  if (idx < BSE) {
    xb[idx] = f2bf(x[idx]);
  } else if (idx < BSE + 4 * (size_t)EE) {
    size_t t = idx - BSE;
    int m = (int)(t / EE);
    int q = (int)(t - (size_t)m * EE);
    int n = q / Eq, k = q - n * Eq;
    const float* W = (m == 0) ? Wqm : (m == 1) ? Wkm : (m == 2) ? Wvm : Wom;
    Wt[t] = f2bf(W[(size_t)k * Eq + n]);
  }
}

// ---------------------------------------------------------------------------
// Kernel 1: fused Q/K/V projection. Each wave computes a 16x64 output strip
// (4 tiles) so the A fragment is loaded once and reused 4x per k-step.
// m = 0 -> Q (row-major), 1 -> K (row-major), 2 -> V stored transposed.
// ---------------------------------------------------------------------------
__global__ void qkv_proj_kernel(const __bf16* __restrict__ xb,
                                const __bf16* __restrict__ Wt,   // q,k,v packed
                                const float* __restrict__ bqm,
                                const float* __restrict__ bkm,
                                const float* __restrict__ bvm,
                                __bf16* __restrict__ Q, __bf16* __restrict__ K,
                                __bf16* __restrict__ Vt) {
  const int wave = blockIdx.x * 4 + (threadIdx.x >> 5);
  const int lane = threadIdx.x & 31;
  const int r = lane & 15, hi = lane >> 4;

  const int stripsPerMat = (Bq * Sq / 16) * (Eq / 64);  // 1024 * 12 = 12288
  const int m   = wave / stripsPerMat;                  // 0..2
  const int rem = wave - m * stripsPerMat;
  const int rt  = rem / (Eq / 64);
  const int cg  = rem - rt * (Eq / 64);
  const int row0 = rt * 16, col0 = cg * 64;

  const __bf16* Wm = Wt + (size_t)m * EE;               // [n][k] bf16
  const float* bias = (m == 0) ? bqm : (m == 1) ? bkm : bvm;

  v8f acc[4] = {};
  for (int k0 = 0; k0 < Eq; k0 += 32) {
    v16bf a = ld_frag_bf16(xb + (size_t)row0 * Eq + k0, Eq, r, hi);
#pragma unroll
    for (int c = 0; c < 4; ++c) {
      v16bf bf = ld_frag_bf16(Wm + (size_t)(col0 + c * 16) * Eq + k0, Eq, r, hi);
      acc[c] = __builtin_amdgcn_wmma_f32_16x16x32_bf16(false, a, false, bf,
                                                       (short)0, acc[c], false, false);
    }
  }

#pragma unroll
  for (int c = 0; c < 4; ++c) {
    union { v8f v; float f[8]; } o; o.v = acc[c];
    const int e = col0 + c * 16 + r;
    const float bia = bias[e];
    if (m == 2) {
#pragma unroll
      for (int p = 0; p < 8; ++p) {
        int gr = row0 + p + 8 * hi;               // global row in [B*S)
        int bb = gr >> 11, s = gr & (Sq - 1);
        Vt[((size_t)bb * Eq + e) * Sq + s] = f2bf(o.f[p] + bia);
      }
    } else {
      __bf16* dst = (m == 0) ? Q : K;
#pragma unroll
      for (int p = 0; p < 8; ++p) {
        int gr = row0 + p + 8 * hi;
        dst[(size_t)gr * Eq + e] = f2bf(o.f[p] + bia);
      }
    }
  }
}

// ---------------------------------------------------------------------------
// Kernel 2: fused flash attention. Grid (S/16, B), 128 threads (4 waves).
// Q row-block staged in LDS; key loop in blocks of 64 (wave w computes the
// 16x16 score subtile for keys kb + w*16). Softmax state in LDS; each wave
// accumulates a 192-wide slice of the 768-wide context in 12 f32 tiles.
// ---------------------------------------------------------------------------
__global__ void flash_attn_kernel(const __bf16* __restrict__ Q,
                                  const __bf16* __restrict__ K,
                                  const __bf16* __restrict__ Vt,
                                  __bf16* __restrict__ CTX) {
  __shared__ __bf16 sQ[16 * Eq];     // 24 KB: whole query row-block
  __shared__ float  sS[16][64];      // raw scores for 16q x 64k
  __shared__ __bf16 sPb[16][64];     // exp(s - m) in bf16 (WMMA A operand)
  __shared__ float  sM[16], sL[16], sAlpha[16];

  const int tid  = threadIdx.x;
  const int w    = tid >> 5;
  const int lane = tid & 31;
  const int r = lane & 15, hi = lane >> 4;
  const int qBase = blockIdx.x * 16;
  const int b     = blockIdx.y;
  const float scale = 0.036084391824351615f;     // 1/sqrt(768)

  const __bf16* Qb  = Q  + ((size_t)b * Sq + qBase) * Eq;
  const __bf16* Kb  = K  + (size_t)b * Sq * Eq;
  const __bf16* Vtb = Vt + (size_t)b * Eq * Sq;

  // Stage Q row-block (contiguous 24576 bytes) into LDS as uint4s.
  {
    const uint4* src = reinterpret_cast<const uint4*>(Qb);
    uint4* dst = reinterpret_cast<uint4*>(sQ);
#pragma unroll
    for (int j = 0; j < 12; ++j)           // 1536 uint4 / 128 threads
      dst[tid + j * 128] = src[tid + j * 128];
  }

  union { v8f v; float f[8]; } O[12];
#pragma unroll
  for (int t = 0; t < 12; ++t) O[t].v = (v8f){};

  if (tid < 16) { sM[tid] = -1e30f; sL[tid] = 0.0f; }
  __syncthreads();

  for (int kb = 0; kb < Sq; kb += 64) {
    // ---- scores: S[q][kb + w*16 + n] over full E ----
    v8f acc = {};
    const __bf16* Krow = Kb + (size_t)(kb + w * 16) * Eq;
#pragma unroll 4
    for (int e0 = 0; e0 < Eq; e0 += 32) {
      v16bf a  = ld_frag_bf16(sQ + e0, Eq, r, hi);        // LDS
      v16bf bf = ld_frag_bf16(Krow + e0, Eq, r, hi);      // global (L2)
      acc = __builtin_amdgcn_wmma_f32_16x16x32_bf16(false, a, false, bf,
                                                    (short)0, acc, false, false);
    }
    union { v8f v; float f[8]; } sc; sc.v = acc;
#pragma unroll
    for (int p = 0; p < 8; ++p)
      sS[p + 8 * hi][w * 16 + r] = sc.f[p] * scale;
    __syncthreads();

    // ---- streaming softmax (one thread per query row) ----
    if (tid < 16) {
      float mOld = sM[tid], mNew = mOld;
#pragma unroll 8
      for (int j = 0; j < 64; ++j) mNew = fmaxf(mNew, sS[tid][j]);
      float alpha = __expf(mOld - mNew);
      float lsum = 0.0f;
#pragma unroll 8
      for (int j = 0; j < 64; ++j) {
        float p = __expf(sS[tid][j] - mNew);
        sPb[tid][j] = f2bf(p);
        lsum += p;
      }
      sM[tid] = mNew;
      sL[tid] = sL[tid] * alpha + lsum;
      sAlpha[tid] = alpha;
    }
    __syncthreads();

    // ---- rescale running context, accumulate P @ V ----
    float al[8];
#pragma unroll
    for (int p = 0; p < 8; ++p) al[p] = sAlpha[p + 8 * hi];
#pragma unroll
    for (int t = 0; t < 12; ++t)
#pragma unroll
      for (int p = 0; p < 8; ++p) O[t].f[p] *= al[p];

    v16bf pa0 = ld_frag_bf16(&sPb[0][0], 64, r, hi);        // K slice 0..31
    v16bf pa1 = ld_frag_bf16(&sPb[0][0] + 32, 64, r, hi);   // K slice 32..63
#pragma unroll
    for (int t = 0; t < 12; ++t) {
      int e_n = w * 192 + t * 16;
      const __bf16* Vtile = Vtb + (size_t)e_n * Sq + kb;    // [e][s], s contiguous
      v16bf b0 = ld_frag_bf16(Vtile, Sq, r, hi);
      v16bf b1 = ld_frag_bf16(Vtile + 32, Sq, r, hi);
      O[t].v = __builtin_amdgcn_wmma_f32_16x16x32_bf16(false, pa0, false, b0,
                                                       (short)0, O[t].v, false, false);
      O[t].v = __builtin_amdgcn_wmma_f32_16x16x32_bf16(false, pa1, false, b1,
                                                       (short)0, O[t].v, false, false);
    }
    __syncthreads();   // protect sS/sPb before next iteration overwrites
  }

  if (tid < 16) sAlpha[tid] = 1.0f / sL[tid];
  __syncthreads();

  float linv[8];
#pragma unroll
  for (int p = 0; p < 8; ++p) linv[p] = sAlpha[p + 8 * hi];
#pragma unroll
  for (int t = 0; t < 12; ++t) {
    int e_n = w * 192 + t * 16 + r;
#pragma unroll
    for (int p = 0; p < 8; ++p) {
      size_t row = (size_t)b * Sq + qBase + p + 8 * hi;
      CTX[row * Eq + e_n] = f2bf(O[t].f[p] * linv[p]);
    }
  }
}

// ---------------------------------------------------------------------------
// Kernel 3: output projection ctx(bf16) @ Wto(bf16) + bo -> fp32 out.
// Each wave computes a 16x64 strip (4 tiles), A fragment reused 4x.
// ---------------------------------------------------------------------------
__global__ void out_proj_kernel(const __bf16* __restrict__ CTX,
                                const __bf16* __restrict__ Wto,  // [n][k] bf16
                                const float* __restrict__ bo,
                                float* __restrict__ out) {
  const int wave = blockIdx.x * 4 + (threadIdx.x >> 5);
  const int lane = threadIdx.x & 31;
  const int r = lane & 15, hi = lane >> 4;

  const int rt = wave / (Eq / 64);
  const int cg = wave - rt * (Eq / 64);
  const int row0 = rt * 16, col0 = cg * 64;

  v8f acc[4] = {};
  for (int k0 = 0; k0 < Eq; k0 += 32) {
    v16bf a = ld_frag_bf16(CTX + (size_t)row0 * Eq + k0, Eq, r, hi);
#pragma unroll
    for (int c = 0; c < 4; ++c) {
      v16bf bf = ld_frag_bf16(Wto + (size_t)(col0 + c * 16) * Eq + k0, Eq, r, hi);
      acc[c] = __builtin_amdgcn_wmma_f32_16x16x32_bf16(false, a, false, bf,
                                                       (short)0, acc[c], false, false);
    }
  }

#pragma unroll
  for (int c = 0; c < 4; ++c) {
    union { v8f v; float f[8]; } o; o.v = acc[c];
    const int e = col0 + c * 16 + r;
    const float bia = bo[e];
#pragma unroll
    for (int p = 0; p < 8; ++p) {
      int gr = row0 + p + 8 * hi;
      out[(size_t)gr * Eq + e] = o.f[p] + bia;
    }
  }
}

// ---------------------------------------------------------------------------
extern "C" void kernel_launch(void* const* d_in, const int* in_sizes, int n_in,
                              void* d_out, int out_size, void* d_ws, size_t ws_size,
                              hipStream_t stream) {
  const float* x  = (const float*)d_in[0];
  const float* Wq = (const float*)d_in[1];
  const float* bq = (const float*)d_in[2];
  const float* Wk = (const float*)d_in[3];
  const float* bk = (const float*)d_in[4];
  const float* Wv = (const float*)d_in[5];
  const float* bv = (const float*)d_in[6];
  const float* Wo = (const float*)d_in[7];
  const float* bo = (const float*)d_in[8];
  float* out = (float*)d_out;

  const size_t BSE = (size_t)Bq * Sq * Eq;       // 12,582,912 elements
  __bf16* Q   = (__bf16*)d_ws;
  __bf16* K   = Q   + BSE;
  __bf16* Vt  = K   + BSE;
  __bf16* CTX = Vt  + BSE;
  __bf16* XB  = CTX + BSE;
  __bf16* WT  = XB  + BSE;            // 4 * E*E: Wtq, Wtk, Wtv, Wto

  // 0) one-time bf16 conversion + weight transposition
  {
    size_t total = BSE + 4 * (size_t)EE;
    int blocks = (int)((total + 255) / 256);
    prep_kernel<<<blocks, 256, 0, stream>>>(x, Wq, Wk, Wv, Wo, XB, WT);
  }
  // 1) QKV projection: 3 * (B*S/16) * (E/64) strips, 4 waves per block
  {
    int strips = 3 * (Bq * Sq / 16) * (Eq / 64);   // 36864
    qkv_proj_kernel<<<strips / 4, 128, 0, stream>>>(XB, WT, bq, bk, bv,
                                                    Q, K, Vt);
  }
  // 2) fused attention
  {
    dim3 grid(Sq / 16, Bq);
    flash_attn_kernel<<<grid, 128, 0, stream>>>(Q, K, Vt, CTX);
  }
  // 3) output projection
  {
    int strips = (Bq * Sq / 16) * (Eq / 64);       // 12288
    out_proj_kernel<<<strips / 4, 128, 0, stream>>>(CTX, WT + 3 * (size_t)EE, bo, out);
  }
}
